// NonLocalBlock_9775345565920
// MI455X (gfx1250) — matrix operand order repaired and checked by reading, not compiled, not run on hardware
//
#include <hip/hip_runtime.h>

typedef __attribute__((ext_vector_type(8)))  __bf16 v8bf;
typedef __attribute__((ext_vector_type(16))) __bf16 v16bf;
typedef __attribute__((ext_vector_type(8)))  float  v8f;

#define LDSK 40  // 32 K elements + 8 pad (80B row stride, keeps 16B alignment)

// Async DMA of 16 bytes global -> LDS (CDNA5 GLOBAL_LOAD_ASYNC_TO_LDS_B128).
// Low 32 bits of a generic shared pointer ARE the LDS offset on AMDGPU.
__device__ __forceinline__ void async_copy16(void* lds, const void* gptr) {
  unsigned int l = (unsigned int)(uintptr_t)lds;
  asm volatile("global_load_async_to_lds_b128 %0, %1, off"
               :
               : "v"(l), "v"(gptr)
               : "memory");
}

// ---------------------------------------------------------------------------
// Generic bf16 WMMA GEMM:  D[M][N] = alpha * A[M][K] (lda) * B[N][K]^T (ldb)
// OUT_MODE 0: bf16 D[m*ldd+n] (+bias[m])
// OUT_MODE 1: bf16 D[n*ldd+m] (+bias[m])   (transposed, packed 16B stores)
// OUT_MODE 2: f32  D[m*ldd+n] * alpha
// OUT_MODE 3: f32  D[m*ldd+n] + bias[m] + resid[m*ldd+n]
// Block tile 128x128, K-step 32, 256 threads = 8 waves (2 M x 4 N),
// each wave computes 64x32 = 8 accumulator tiles of 16x16.
// Double-buffered LDS filled by async-to-LDS DMA (ASYNCcnt pipeline).
// ---------------------------------------------------------------------------
template <int OUT_MODE, bool HAS_BIAS, bool HAS_RESID>
__global__ __launch_bounds__(256) void gemm_bf16_wmma(
    const __bf16* __restrict__ A, const __bf16* __restrict__ B,
    void* __restrict__ Dvoid, const float* __restrict__ bias,
    const float* __restrict__ resid, int K, int lda, int ldb, int ldd,
    long long sA, long long sB, long long sD, long long sR, float alpha) {
  const int bz = blockIdx.z;
  A += (long long)bz * sA;
  B += (long long)bz * sB;

  const int tid   = threadIdx.x;
  const int wave  = tid >> 5;
  const int lane  = tid & 31;
  const int waveM = wave >> 2;  // 0..1
  const int waveN = wave & 3;   // 0..3
  const int bm0   = blockIdx.y * 128;
  const int bn0   = blockIdx.x * 128;
  const int llo   = lane & 15;
  const int lhi   = lane >> 4;

  __shared__ __align__(16) __bf16 As[2][128 * LDSK];
  __shared__ __align__(16) __bf16 Bs[2][128 * LDSK];

  // Per-thread staging chunks: chunk ids tid and tid+256 cover 128x32 bf16.
  const int row0 = tid >> 2;                  // 0..63
  const int kc0  = (tid & 3) * 8;             // 0,8,16,24
  const int row1 = (tid + 256) >> 2;          // 64..127
  const int kc1  = kc0;

  v8f acc[4][2] = {};

  const int KT = K >> 5;  // K-steps of 32

  // Prologue: stage tile 0 into buffer 0 (4 async instrs per wave).
  {
    const int k0 = 0;
    async_copy16(&As[0][row0 * LDSK + kc0],
                 A + (size_t)(bm0 + row0) * lda + k0 + kc0);
    async_copy16(&As[0][row1 * LDSK + kc1],
                 A + (size_t)(bm0 + row1) * lda + k0 + kc1);
    async_copy16(&Bs[0][row0 * LDSK + kc0],
                 B + (size_t)(bn0 + row0) * ldb + k0 + kc0);
    async_copy16(&Bs[0][row1 * LDSK + kc1],
                 B + (size_t)(bn0 + row1) * ldb + k0 + kc1);
  }

  for (int kt = 0; kt < KT; ++kt) {
    const int cur = kt & 1;
    if (kt + 1 < KT) {
      const int nxt = cur ^ 1;
      const int k0  = (kt + 1) << 5;
      async_copy16(&As[nxt][row0 * LDSK + kc0],
                   A + (size_t)(bm0 + row0) * lda + k0 + kc0);
      async_copy16(&As[nxt][row1 * LDSK + kc1],
                   A + (size_t)(bm0 + row1) * lda + k0 + kc1);
      async_copy16(&Bs[nxt][row0 * LDSK + kc0],
                   B + (size_t)(bn0 + row0) * ldb + k0 + kc0);
      async_copy16(&Bs[nxt][row1 * LDSK + kc1],
                   B + (size_t)(bn0 + row1) * ldb + k0 + kc1);
      // 8 outstanding; wait until only the 4 just-issued remain.
      asm volatile("s_wait_asynccnt 0x4" ::: "memory");
    } else {
      asm volatile("s_wait_asynccnt 0x0" ::: "memory");
    }
    __syncthreads();

    // A fragment (16x32 MxK): lanes 0-15 K=0..7,16..23; lanes 16-31 K=8..15,24..31
    v16bf afrag[4];
#pragma unroll
    for (int mt = 0; mt < 4; ++mt) {
      const int r  = waveM * 64 + mt * 16 + llo;
      const int kb = lhi * 8;
      v8bf lo = *(const v8bf*)&As[cur][r * LDSK + kb];
      v8bf hi = *(const v8bf*)&As[cur][r * LDSK + kb + 16];
      afrag[mt] = __builtin_shufflevector(lo, hi, 0, 1, 2, 3, 4, 5, 6, 7, 8, 9,
                                          10, 11, 12, 13, 14, 15);
    }
    // B fragment (32x16 KxN): lanes 0-15 K=0..7,16..23; lanes 16-31 K=8..15,24..31
    v16bf bfrag[2];
#pragma unroll
    for (int nt = 0; nt < 2; ++nt) {
      const int r  = waveN * 32 + nt * 16 + llo;
      const int kb = lhi * 8;
      v8bf lo = *(const v8bf*)&Bs[cur][r * LDSK + kb];
      v8bf hi = *(const v8bf*)&Bs[cur][r * LDSK + kb + 16];
      bfrag[nt] = __builtin_shufflevector(lo, hi, 0, 1, 2, 3, 4, 5, 6, 7, 8, 9,
                                          10, 11, 12, 13, 14, 15);
    }
#pragma unroll
    for (int mt = 0; mt < 4; ++mt)
#pragma unroll
      for (int nt = 0; nt < 2; ++nt)
        acc[mt][nt] = __builtin_amdgcn_wmma_f32_16x16x32_bf16(
            false, afrag[mt], false, bfrag[nt], (short)0, acc[mt][nt], false,
            false);
    __syncthreads();
  }

  // Epilogue. C/D layout: VGPR r -> M = base + r + 8*lhi, N = base + llo.
#pragma unroll
  for (int mt = 0; mt < 4; ++mt) {
#pragma unroll
    for (int nt = 0; nt < 2; ++nt) {
      const int m0 = bm0 + waveM * 64 + mt * 16 + lhi * 8;
      const int n  = bn0 + waveN * 32 + nt * 16 + llo;
      v8f c = acc[mt][nt];
      if constexpr (OUT_MODE == 0) {
        __bf16* D = (__bf16*)Dvoid + (long long)bz * sD;
#pragma unroll
        for (int r = 0; r < 8; ++r) {
          float v = c[r];
          if constexpr (HAS_BIAS) v += bias[m0 + r];
          D[(size_t)(m0 + r) * ldd + n] = (__bf16)v;
        }
      } else if constexpr (OUT_MODE == 1) {
        __bf16* D = (__bf16*)Dvoid + (long long)bz * sD;
        union { uint4 u; __bf16 h[8]; } P;
#pragma unroll
        for (int r = 0; r < 8; ++r) {
          float v = c[r];
          if constexpr (HAS_BIAS) v += bias[m0 + r];
          P.h[r] = (__bf16)v;
        }
        *(uint4*)&D[(size_t)n * ldd + m0] = P.u;  // m0 multiple of 8 -> 16B aligned
      } else if constexpr (OUT_MODE == 2) {
        float* D = (float*)Dvoid + (long long)bz * sD;
#pragma unroll
        for (int r = 0; r < 8; ++r)
          D[(size_t)(m0 + r) * ldd + n] = c[r] * alpha;
      } else {
        float* D = (float*)Dvoid + (long long)bz * sD;
#pragma unroll
        for (int r = 0; r < 8; ++r) {
          float v = c[r];
          if constexpr (HAS_BIAS) v += bias[m0 + r];
          if constexpr (HAS_RESID)
            v += resid[(long long)bz * sR + (size_t)(m0 + r) * ldd + n];
          D[(size_t)(m0 + r) * ldd + n] = v;
        }
      }
    }
  }
}

// ---- fp32 weights -> bf16 (4 matrices of 512x512, packed [theta,phi,g,out])
__global__ void cvt_weights(const float* __restrict__ wt,
                            const float* __restrict__ wp,
                            const float* __restrict__ wg,
                            const float* __restrict__ wo,
                            __bf16* __restrict__ Wb) {
  const int i = blockIdx.x * 256 + threadIdx.x;  // 0..262143
  Wb[i]              = (__bf16)wt[i];
  Wb[262144 + i]     = (__bf16)wp[i];
  Wb[2 * 262144 + i] = (__bf16)wg[i];
  Wb[3 * 262144 + i] = (__bf16)wo[i];
}

// ---- x[b][c][n] f32 -> xT[b][n][c] bf16 via 32x32 LDS tiles
__global__ void x_transpose_bf16(const float* __restrict__ x,
                                 __bf16* __restrict__ xT) {
  const int C = 512, N = 4096;
  __shared__ float t[32][33];
  const int b  = blockIdx.z;
  const int n0 = blockIdx.x * 32;
  const int c0 = blockIdx.y * 32;
  const int tx = threadIdx.x, ty = threadIdx.y;  // (32,8)
  const float* xb = x + (size_t)b * C * N;
  __bf16* xo = xT + (size_t)b * N * C;
#pragma unroll
  for (int i = 0; i < 32; i += 8)
    t[ty + i][tx] = xb[(size_t)(c0 + ty + i) * N + n0 + tx];
  __syncthreads();
#pragma unroll
  for (int i = 0; i < 32; i += 8)
    xo[(size_t)(n0 + ty + i) * C + c0 + tx] = (__bf16)t[tx][ty + i];
}

// ---- row softmax over 512 f32 -> bf16; one 256-thread block per row
__global__ void softmax_rows(const float* __restrict__ S,
                             __bf16* __restrict__ Aout) {
  const int cols = 512;
  const long long row = blockIdx.x;
  const float* s = S + row * cols;
  __bf16* a = Aout + row * cols;
  const int tid = threadIdx.x, wave = tid >> 5, lane = tid & 31;
  __shared__ float sm[8], ss[8];

  float v0 = s[tid], v1 = s[tid + 256];
  float m = fmaxf(v0, v1);
#pragma unroll
  for (int o = 16; o > 0; o >>= 1) m = fmaxf(m, __shfl_xor(m, o, 32));
  if (lane == 0) sm[wave] = m;
  __syncthreads();
  if (wave == 0) {
    float t = (lane < 8) ? sm[lane] : -3.4e38f;
#pragma unroll
    for (int o = 4; o > 0; o >>= 1) t = fmaxf(t, __shfl_xor(t, o, 32));
    if (lane == 0) sm[0] = t;
  }
  __syncthreads();
  const float rmax = sm[0];
  float e0 = __expf(v0 - rmax), e1 = __expf(v1 - rmax);
  float su = e0 + e1;
#pragma unroll
  for (int o = 16; o > 0; o >>= 1) su += __shfl_xor(su, o, 32);
  if (lane == 0) ss[wave] = su;
  __syncthreads();
  if (wave == 0) {
    float t = (lane < 8) ? ss[lane] : 0.0f;
#pragma unroll
    for (int o = 4; o > 0; o >>= 1) t += __shfl_xor(t, o, 32);
    if (lane == 0) ss[0] = t;
  }
  __syncthreads();
  const float inv = 1.0f / ss[0];
  a[tid]       = (__bf16)(e0 * inv);
  a[tid + 256] = (__bf16)(e1 * inv);
}

extern "C" void kernel_launch(void* const* d_in, const int* in_sizes, int n_in,
                              void* d_out, int out_size, void* d_ws,
                              size_t ws_size, hipStream_t stream) {
  const float* x  = (const float*)d_in[0];
  const float* Wt = (const float*)d_in[1];
  const float* bt = (const float*)d_in[2];
  const float* Wp = (const float*)d_in[3];
  const float* bp = (const float*)d_in[4];
  const float* Wg = (const float*)d_in[5];
  const float* bg = (const float*)d_in[6];
  const float* Wo = (const float*)d_in[7];
  const float* bo = (const float*)d_in[8];
  float* out = (float*)d_out;

  const int Bn = 16, C = 512, N = 4096;
  const long long actE = (long long)Bn * C * N;  // elements per activation

  char* ws = (char*)d_ws;
  __bf16* xT    = (__bf16*)ws;                    // [B][N][C], later reused as yT
  __bf16* phi   = (__bf16*)(ws + actE * 2);       // [B][C][N]
  __bf16* theta = (__bf16*)(ws + actE * 4);       // [B][C][N]
  __bf16* gT    = (__bf16*)(ws + actE * 6);       // [B][N][C]
  float*  sc    = (float*)(ws + actE * 8);        // [B][C][C] f32
  __bf16* attn  = (__bf16*)(ws + actE * 8 + (long long)Bn * C * C * 4);
  __bf16* Wb    = (__bf16*)(ws + actE * 8 + (long long)Bn * C * C * 6);
  __bf16* yT    = xT;  // xT dead after QKV GEMMs

  const long long sAct  = (long long)C * N;  // per-batch activation stride
  const long long sAttn = (long long)C * C;

  cvt_weights<<<1024, 256, 0, stream>>>(Wt, Wp, Wg, Wo, Wb);
  x_transpose_bf16<<<dim3(128, 16, 16), dim3(32, 8), 0, stream>>>(x, xT);

  // phi = W_phi * x + b_phi   -> [c][n]
  gemm_bf16_wmma<0, true, false><<<dim3(32, 4, 16), 256, 0, stream>>>(
      Wb + 262144, xT, phi, bp, nullptr, C, C, C, N, 0, sAct, sAct, 0, 1.0f);
  // theta = W_theta * x + b_theta -> [c][n]
  gemm_bf16_wmma<0, true, false><<<dim3(32, 4, 16), 256, 0, stream>>>(
      Wb, xT, theta, bt, nullptr, C, C, C, N, 0, sAct, sAct, 0, 1.0f);
  // g = W_g * x + b_g -> stored transposed gT[n][c]
  gemm_bf16_wmma<1, true, false><<<dim3(32, 4, 16), 256, 0, stream>>>(
      Wb + 2 * 262144, xT, gT, bg, nullptr, C, C, C, C, 0, sAct, sAct, 0, 1.0f);

  // scores[c][d] = (1/C) * sum_n phi[c][n] * theta[d][n]
  gemm_bf16_wmma<2, false, false><<<dim3(4, 4, 16), 256, 0, stream>>>(
      phi, theta, sc, nullptr, nullptr, N, N, N, C, sAct, sAct, sAttn, 0,
      1.0f / 512.0f);

  softmax_rows<<<Bn * C, 256, 0, stream>>>(sc, attn);

  // yT[n][c] = sum_d attn[c][d] * gT[n][d]
  gemm_bf16_wmma<1, false, false><<<dim3(32, 4, 16), 256, 0, stream>>>(
      attn, gT, yT, nullptr, nullptr, C, C, C, C, sAttn, sAct, sAct, 0, 1.0f);

  // out = x + W_out * Y + b_out
  gemm_bf16_wmma<3, true, true><<<dim3(32, 4, 16), 256, 0, stream>>>(
      Wb + 3 * 262144, yT, out, bo, x, C, C, C, N, 0, sAct, sAct, sAct, 1.0f);

  (void)in_sizes; (void)n_in; (void)out_size; (void)ws_size;
}
